// Attention_1090921693884
// MI455X (gfx1250) — matrix operand order repaired and checked by reading, not compiled, not run on hardware
//
#include <hip/hip_runtime.h>

typedef unsigned short ushort_t;
typedef __attribute__((ext_vector_type(16))) __bf16 v16bf;
typedef __attribute__((ext_vector_type(8)))  float  v8f;

#define B_    4
#define S_    2048
#define D_    1024
#define H_    16
#define HD_   64
#define HALF_ 32

union FragAB {
    v16bf v;
    uint4 q[2];
};

__device__ __forceinline__ ushort_t f2bf(float x) {
    unsigned int u = __float_as_uint(x);
    u += 0x7FFFu + ((u >> 16) & 1u);   // round-to-nearest-even
    return (ushort_t)(u >> 16);
}

// A-matrix 16x32 bf16 fragment (M x K). base points at (row0, k0); rows have
// `stride` elements, K contiguous. Lane L<16: M=L, K={0..7,16..23};
// lane L>=16: M=L-16, K={8..15,24..31}.  (ISA 7.12.2)
__device__ __forceinline__ void loadFragA(const ushort_t* base, int stride,
                                          FragAB& f, int lane) {
    int row = lane & 15;
    int kb  = (lane >> 4) << 3;            // 0 or 8
    const ushort_t* p = base + row * stride + kb;
    f.q[0] = *reinterpret_cast<const uint4*>(p);       // K = kb..kb+7
    f.q[1] = *reinterpret_cast<const uint4*>(p + 16);  // K = kb+16..kb+23
}

// B-matrix 32x16 bf16 fragment (K x N), source stored "one row per n,
// contiguous in k" (i.e. element (k,n) at base + n*stride + k).
// Lane L<16: N=L, K={0..7,16..23}; lane L>=16: N=L-16, K={8..15,24..31}.
__device__ __forceinline__ void loadFragB(const ushort_t* base, int stride,
                                          FragAB& f, int lane) {
    int n  = lane & 15;
    int kb = (lane >> 4) << 3;             // 0 or 8
    const ushort_t* p = base + n * stride + kb;
    f.q[0] = *reinterpret_cast<const uint4*>(p);       // K = kb..kb+7
    f.q[1] = *reinterpret_cast<const uint4*>(p + 16);  // K = kb+16..kb+23
}

__device__ __forceinline__ v8f wmma_bf16(const FragAB& a, const FragAB& b, v8f c) {
    return __builtin_amdgcn_wmma_f32_16x16x32_bf16(
        /*neg_a=*/false, a.v, /*neg_b=*/false, b.v,
        /*c_mod=*/(short)0, c, /*reuse_a=*/false, /*reuse_b=*/false);
}

// ---------------------------------------------------------------- fp32->bf16
__global__ __launch_bounds__(256) void cvt_bf16_kernel(const float* __restrict__ in,
                                                       ushort_t* __restrict__ out,
                                                       int n) {
    int i = (blockIdx.x * 256 + threadIdx.x) * 4;
    if (i < n) {
        float4 f = *reinterpret_cast<const float4*>(in + i);
        unsigned long long pack =
            (unsigned long long)f2bf(f.x) |
            ((unsigned long long)f2bf(f.y) << 16) |
            ((unsigned long long)f2bf(f.z) << 32) |
            ((unsigned long long)f2bf(f.w) << 48);
        *reinterpret_cast<unsigned long long*>(out + i) = pack;
    }
}

// ------------------------------------------------- fused QKV GEMM + RoPE
// grid: (BS/512, D/64, 3).  8 waves/block, wave tile 64x64 (4x4 wmma tiles):
// 16 wmma per 16 b128 loads per K-step, 4x register reuse of each B fragment.
// out = x @ W^T (NT GEMM, both operands K-major -> direct WMMA fragments).
__global__ __launch_bounds__(256) void qkv_rope_kernel(
    const ushort_t* __restrict__ xbf,
    const ushort_t* __restrict__ wqbf,
    const ushort_t* __restrict__ wkbf,
    const ushort_t* __restrict__ wvbf,
    const float* __restrict__ cosT,
    const float* __restrict__ sinT,
    ushort_t* __restrict__ qT,   // [B,H,S,HD]
    ushort_t* __restrict__ kT,   // [B,H,S,HD]
    ushort_t* __restrict__ vT)   // [B,H,HD,S]  (transposed for P@V)
{
    int tid  = threadIdx.x;
    int wave = tid >> 5;
    int lane = tid & 31;
    int m0   = blockIdx.x * 512 + wave * 64;
    int n0   = blockIdx.y * 64;
    int wsel = blockIdx.z;
    const ushort_t* W = (wsel == 0) ? wqbf : (wsel == 1) ? wkbf : wvbf;

    v8f acc[4][4];
#pragma unroll
    for (int mi = 0; mi < 4; ++mi)
#pragma unroll
        for (int nt = 0; nt < 4; ++nt)
#pragma unroll
            for (int e = 0; e < 8; ++e) acc[mi][nt][e] = 0.0f;

    for (int k = 0; k < D_; k += 32) {
        FragAB a[4], bf;
#pragma unroll
        for (int mi = 0; mi < 4; ++mi)
            loadFragA(xbf + (m0 + mi * 16) * D_ + k, D_, a[mi], lane);
#pragma unroll
        for (int nt = 0; nt < 4; ++nt) {
            loadFragB(W + (n0 + nt * 16) * D_ + k, D_, bf, lane);
#pragma unroll
            for (int mi = 0; mi < 4; ++mi)
                acc[mi][nt] = wmma_bf16(a[mi], bf, acc[mi][nt]);
        }
    }

    // Epilogue: RoPE (q,k) via lane-pair shuffle, scatter to attention layouts.
    int hiM      = (lane >> 4) << 3;
    int colLane  = lane & 15;
#pragma unroll
    for (int mi = 0; mi < 4; ++mi)
#pragma unroll
        for (int nt = 0; nt < 4; ++nt)
#pragma unroll
            for (int r = 0; r < 8; ++r) {
                int t = m0 + mi * 16 + hiM + r;      // token index
                int n = n0 + nt * 16 + colLane;      // feature index
                float val  = acc[mi][nt][r];
                int b    = t >> 11;                  // / S_
                int spos = t & (S_ - 1);
                int h    = n >> 6;
                int j    = n & 63;
                float res = val;
                if (wsel < 2) {
                    int   i = j >> 1;
                    float c = cosT[spos * HALF_ + i];
                    float s = sinT[spos * HALF_ + i];
                    float other = __shfl_xor(val, 1);   // partner of RoPE pair
                    res = ((n & 1) == 0) ? (val * c - other * s)
                                         : (other * s + val * c);
                }
                ushort_t bv = f2bf(res);
                int bh = b * H_ + h;
                if (wsel == 0)       qT[(bh * S_ + spos) * HD_ + j] = bv;
                else if (wsel == 1)  kT[(bh * S_ + spos) * HD_ + j] = bv;
                else                 vT[(bh * HD_ + j) * S_ + spos] = bv;
            }
}

// ------------------------------------------------- flash attention (causal)
// grid: (S/128, B*H). 8 waves/block, each wave owns 16 query rows.
__global__ __launch_bounds__(256) void attn_kernel(
    const ushort_t* __restrict__ qT,
    const ushort_t* __restrict__ kT,
    const ushort_t* __restrict__ vT,
    ushort_t* __restrict__ attnO)    // [B,S,D] bf16
{
    __shared__ __align__(16) ushort_t sP[8 * 16 * 72];  // per-wave 16x64 P tile

    int tid  = threadIdx.x;
    int wave = tid >> 5;
    int lane = tid & 31;
    int bh   = blockIdx.y;
    int b    = bh >> 4;
    int h    = bh & 15;
    int m0   = blockIdx.x * 128;
    int q0   = m0 + wave * 16;
    int hiM      = (lane >> 4) << 3;
    int colLane  = lane & 15;

    FragAB qA0, qA1;
    loadFragA(qT + (bh * S_ + q0) * HD_ +  0, HD_, qA0, lane);
    loadFragA(qT + (bh * S_ + q0) * HD_ + 32, HD_, qA1, lane);

    float mrow[8], lrow[8];
    v8f o[4];
#pragma unroll
    for (int r = 0; r < 8; ++r) { mrow[r] = -3.0e38f; lrow[r] = 0.0f; }
#pragma unroll
    for (int ft = 0; ft < 4; ++ft)
#pragma unroll
        for (int e = 0; e < 8; ++e) o[ft][e] = 0.0f;

    ushort_t* myP = sP + wave * 16 * 72;
    int nblk = (q0 >> 6) + 1;                    // causal: keys <= q0+15

    for (int blk = 0; blk < nblk; ++blk) {
        int jb = blk * 64;

        // ---- scores 16x64 = Q(16x64) @ K^T
        v8f sacc[4];
#pragma unroll
        for (int t = 0; t < 4; ++t) {
#pragma unroll
            for (int e = 0; e < 8; ++e) sacc[t][e] = 0.0f;
            FragAB kB;
            loadFragB(kT + (bh * S_ + jb + t * 16) * HD_ +  0, HD_, kB, lane);
            sacc[t] = wmma_bf16(qA0, kB, sacc[t]);
            loadFragB(kT + (bh * S_ + jb + t * 16) * HD_ + 32, HD_, kB, lane);
            sacc[t] = wmma_bf16(qA1, kB, sacc[t]);
        }

        // ---- scale + causal mask + row max (16 lanes of half-wave = one row)
        float mloc[8];
#pragma unroll
        for (int r = 0; r < 8; ++r) mloc[r] = -3.0e38f;
#pragma unroll
        for (int t = 0; t < 4; ++t)
#pragma unroll
            for (int r = 0; r < 8; ++r) {
                int key = jb + t * 16 + colLane;
                int row = q0 + hiM + r;
                float sc = sacc[t][r] * 0.125f;       // 1/sqrt(64)
                sc = (key <= row) ? sc : -3.0e38f;
                sacc[t][r] = sc;
                mloc[r] = fmaxf(mloc[r], sc);
            }
#pragma unroll
        for (int m = 1; m < 16; m <<= 1)
#pragma unroll
            for (int r = 0; r < 8; ++r)
                mloc[r] = fmaxf(mloc[r], __shfl_xor(mloc[r], m));

        float alpha[8];
#pragma unroll
        for (int r = 0; r < 8; ++r) {
            float mnew = fmaxf(mrow[r], mloc[r]);
            alpha[r] = __expf(mrow[r] - mnew);
            mrow[r] = mnew;
        }

        // ---- P = exp(S - m), row sums
        float rsum[8];
#pragma unroll
        for (int r = 0; r < 8; ++r) rsum[r] = 0.0f;
#pragma unroll
        for (int t = 0; t < 4; ++t)
#pragma unroll
            for (int r = 0; r < 8; ++r) {
                float p = __expf(sacc[t][r] - mrow[r]);
                sacc[t][r] = p;
                rsum[r] += p;
            }
#pragma unroll
        for (int m = 1; m < 16; m <<= 1)
#pragma unroll
            for (int r = 0; r < 8; ++r)
                rsum[r] += __shfl_xor(rsum[r], m);
#pragma unroll
        for (int r = 0; r < 8; ++r) lrow[r] = lrow[r] * alpha[r] + rsum[r];

        // ---- rescale running output
#pragma unroll
        for (int ft = 0; ft < 4; ++ft)
#pragma unroll
            for (int r = 0; r < 8; ++r) o[ft][r] *= alpha[r];

        // ---- C-layout -> A-layout conversion of P through wave-private LDS
#pragma unroll
        for (int t = 0; t < 4; ++t)
#pragma unroll
            for (int r = 0; r < 8; ++r)
                myP[(hiM + r) * 72 + t * 16 + colLane] = f2bf(sacc[t][r]);
        asm volatile("s_wait_dscnt 0" ::: "memory");   // wave-local DS ordering

        FragAB pA0, pA1;
        loadFragA(myP +  0, 72, pA0, lane);
        loadFragA(myP + 32, 72, pA1, lane);

        // ---- O += P(16x64) @ V(64x64) ; vT rows contiguous in key dim
#pragma unroll
        for (int ft = 0; ft < 4; ++ft) {
            FragAB vB;
            loadFragB(vT + (bh * HD_ + ft * 16) * S_ + jb +  0, S_, vB, lane);
            o[ft] = wmma_bf16(pA0, vB, o[ft]);
            loadFragB(vT + (bh * HD_ + ft * 16) * S_ + jb + 32, S_, vB, lane);
            o[ft] = wmma_bf16(pA1, vB, o[ft]);
        }
    }

    // ---- normalize + store [B,S,D]
#pragma unroll
    for (int r = 0; r < 8; ++r) lrow[r] = 1.0f / lrow[r];
#pragma unroll
    for (int ft = 0; ft < 4; ++ft)
#pragma unroll
        for (int r = 0; r < 8; ++r) {
            int row = q0 + hiM + r;
            int col = h * HD_ + ft * 16 + colLane;
            attnO[(b * S_ + row) * D_ + col] = f2bf(o[ft][r] * lrow[r]);
        }
}

// ------------------------------------------------- output projection
// grid: (BS/512, D/64).  out[t,n] = attnO[t,:] . wo[n,:]   (fp32 out)
// Same 64x64 wave tile as the QKV GEMM.
__global__ __launch_bounds__(256) void proj_kernel(
    const ushort_t* __restrict__ aO,
    const ushort_t* __restrict__ wobf,
    float* __restrict__ out)
{
    int tid  = threadIdx.x;
    int wave = tid >> 5;
    int lane = tid & 31;
    int m0   = blockIdx.x * 512 + wave * 64;
    int n0   = blockIdx.y * 64;

    v8f acc[4][4];
#pragma unroll
    for (int mi = 0; mi < 4; ++mi)
#pragma unroll
        for (int nt = 0; nt < 4; ++nt)
#pragma unroll
            for (int e = 0; e < 8; ++e) acc[mi][nt][e] = 0.0f;

    for (int k = 0; k < D_; k += 32) {
        FragAB a[4], bf;
#pragma unroll
        for (int mi = 0; mi < 4; ++mi)
            loadFragA(aO + (m0 + mi * 16) * D_ + k, D_, a[mi], lane);
#pragma unroll
        for (int nt = 0; nt < 4; ++nt) {
            loadFragB(wobf + (n0 + nt * 16) * D_ + k, D_, bf, lane);
#pragma unroll
            for (int mi = 0; mi < 4; ++mi)
                acc[mi][nt] = wmma_bf16(a[mi], bf, acc[mi][nt]);
        }
    }

    int hiM     = (lane >> 4) << 3;
    int colLane = lane & 15;
#pragma unroll
    for (int mi = 0; mi < 4; ++mi)
#pragma unroll
        for (int nt = 0; nt < 4; ++nt)
#pragma unroll
            for (int r = 0; r < 8; ++r) {
                int t = m0 + mi * 16 + hiM + r;
                int n = n0 + nt * 16 + colLane;
                out[t * D_ + n] = acc[mi][nt][r];
            }
}

extern "C" void kernel_launch(void* const* d_in, const int* in_sizes, int n_in,
                              void* d_out, int out_size, void* d_ws, size_t ws_size,
                              hipStream_t stream) {
    (void)in_sizes; (void)n_in; (void)out_size; (void)ws_size;
    const float* x    = (const float*)d_in[0];
    const float* cosT = (const float*)d_in[1];
    const float* sinT = (const float*)d_in[2];
    const float* wq   = (const float*)d_in[3];
    const float* wk   = (const float*)d_in[4];
    const float* wv   = (const float*)d_in[5];
    const float* wo   = (const float*)d_in[6];
    float* out = (float*)d_out;

    char* ws = (char*)d_ws;
    const size_t MB = 1u << 20;
    ushort_t* xbf  = (ushort_t*)(ws);             // 16 MB (reused as attnO later)
    ushort_t* wqbf = (ushort_t*)(ws + 16 * MB);   //  2 MB
    ushort_t* wkbf = (ushort_t*)(ws + 18 * MB);   //  2 MB
    ushort_t* wvbf = (ushort_t*)(ws + 20 * MB);   //  2 MB
    ushort_t* wobf = (ushort_t*)(ws + 22 * MB);   //  2 MB
    ushort_t* qT   = (ushort_t*)(ws + 24 * MB);   // 16 MB
    ushort_t* kT   = (ushort_t*)(ws + 40 * MB);   // 16 MB
    ushort_t* vT   = (ushort_t*)(ws + 56 * MB);   // 16 MB  (total 72 MB)
    ushort_t* attnO = xbf;                        // x dead after QKV GEMM

    const int NX = B_ * S_ * D_;                  // 8M
    const int NW = D_ * D_;                       // 1M
    cvt_bf16_kernel<<<NX / 4 / 256, 256, 0, stream>>>(x,  xbf,  NX);
    cvt_bf16_kernel<<<NW / 4 / 256, 256, 0, stream>>>(wq, wqbf, NW);
    cvt_bf16_kernel<<<NW / 4 / 256, 256, 0, stream>>>(wk, wkbf, NW);
    cvt_bf16_kernel<<<NW / 4 / 256, 256, 0, stream>>>(wv, wvbf, NW);
    cvt_bf16_kernel<<<NW / 4 / 256, 256, 0, stream>>>(wo, wobf, NW);

    qkv_rope_kernel<<<dim3(B_ * S_ / 512, D_ / 64, 3), 256, 0, stream>>>(
        xbf, wqbf, wkbf, wvbf, cosT, sinT, qT, kT, vT);

    attn_kernel<<<dim3(S_ / 128, B_ * H_), 256, 0, stream>>>(qT, kT, vT, attnO);

    proj_kernel<<<dim3(B_ * S_ / 512, D_ / 64), 256, 0, stream>>>(attnO, wobf, out);
}
